// Net_936302871004
// MI455X (gfx1250) — matrix operand order repaired and checked by reading, not compiled, not run on hardware
//
#include <hip/hip_runtime.h>
#include <cstdint>
#include <cstddef>

typedef __attribute__((ext_vector_type(16))) _Float16 v16h;
typedef __attribute__((ext_vector_type(8)))  float    v8f;

#define USE_ASYNC_LDS 1   // set to 0 to fall back to VGPR-staged LDS copies

namespace {

constexpr int BGRAPH = 16;
constexpr int NNODE  = 4096;
constexpr int NTOT   = BGRAPH * NNODE;     // 65536
constexpr int DEG    = 8;
constexpr int ETOT   = NTOT * DEG * 2;     // 1048576
constexpr int CLS    = 10;
constexpr int KPER   = 2048;
constexpr int KTOT   = BGRAPH * KPER;      // 32768
constexpr unsigned HSIZE = 1u << 21;       // hash slots (>= 2x max unique coarse edges)

// ---------------------------------------------------------------- utilities
__global__ void fill_u32_kernel(uint32_t* __restrict__ p, uint32_t v, int n) {
  int i = blockIdx.x * blockDim.x + threadIdx.x;
  if (i < n) p[i] = v;
}

// Pack fp32 weight W[K][NO] into f16 WMMA B-fragments.
// Fragment f = kstep*NT + ntile is 32 lanes x 8 dwords (contiguous, 1KB):
//   lane L: col = L%16, kbase = (L/16)*16; dword p holds K = kbase+2p, kbase+2p+1.
__global__ void pack_weight_kernel(const float* __restrict__ W, uint32_t* __restrict__ out,
                                   int K, int NO) {
  int d = blockIdx.x * blockDim.x + threadIdx.x;
  int total = (K * NO) >> 1;
  if (d >= total) return;
  int NT = NO >> 4;
  int f = d >> 8;
  int r = d & 255;
  int lane = r >> 3, p = r & 7;
  int kstep = f / NT, ntile = f % NT;
  int col = lane & 15;
  int k0 = kstep * 32 + ((lane >> 4) << 4) + 2 * p;
  _Float16 lo = (_Float16)W[k0 * NO + ntile * 16 + col];
  _Float16 hi = (_Float16)W[(k0 + 1) * NO + ntile * 16 + col];
  unsigned short ulo, uhi;
  __builtin_memcpy(&ulo, &lo, 2);
  __builtin_memcpy(&uhi, &hi, 2);
  out[d] = (uint32_t)ulo | ((uint32_t)uhi << 16);
}

// ------------------------------------------------------------- WMMA GEMM
// Y[M,NO] = act( X1@W1 (+ X2@W2) + bias ), weights pre-packed as B-fragments.
// 256 threads = 8 waves; each wave computes a 16xNO tile; 128 rows per block.

template<int N>
__device__ __forceinline__ void wait_async() {
#if USE_ASYNC_LDS
  asm volatile("s_wait_asynccnt %0" :: "i"(N) : "memory");
#endif
}

#if USE_ASYNC_LDS
// Async copy of a 128xKD fp32 tile into LDS via the CDNA5 async-DMA path
// (GLOBAL_LOAD_ASYNC_TO_LDS_B128, GVS addressing: saddr sgpr-pair + 32b voffset).
// Each thread issues (128*KD)/(256*4) in-order async ops; no VGPR round-trip.
template<int KD>
__device__ __forceinline__ void gemm_stage(float* tile, const float* __restrict__ X,
                                           long rowBase, int tid) {
  const uint64_t gbase = (uint64_t)(uintptr_t)(X + rowBase * KD);
#pragma unroll
  for (int i = 0; i < (128 * KD) / (256 * 4); ++i) {
    int linear = (i * 256 + tid) * 4;
    int r = linear / KD, c = linear % KD;
    uint32_t lds  = (uint32_t)(uintptr_t)(tile + r * (KD + 4) + c);  // addr[31:0] == LDS offset
    uint32_t goff = (uint32_t)(linear * 4);
    asm volatile("global_load_async_to_lds_b128 %0, %1, %2"
                 :: "v"(lds), "v"(goff), "s"(gbase)
                 : "memory");
  }
}
#else
template<int KD>
__device__ __forceinline__ void gemm_stage(float* tile, const float* __restrict__ X,
                                           long rowBase, int tid) {
#pragma unroll
  for (int i = 0; i < (128 * KD) / (256 * 4); ++i) {
    int linear = (i * 256 + tid) * 4;
    int r = linear / KD, c = linear % KD;
    const float4 v = *(const float4*)(X + (rowBase + r) * KD + c);
    *(float4*)(tile + r * (KD + 4) + c) = v;   // pad=4 keeps 16B alignment
  }
}
#endif

template<int KD, int NT>
__device__ __forceinline__ void gemm_phase(const float* tile, const uint32_t* __restrict__ pack,
                                           int lane, int wave, v8f* acc) {
  const int half = lane >> 4;
  const float* trow = tile + (wave * 16 + (lane & 15)) * (KD + 4);
#pragma unroll
  for (int ks = 0; ks < KD / 32; ++ks) {
    // A fragment: 16x32 f16, lane L: row=L%16; pair p -> K = ks*32 + (L/16)*16 + p*2
    v16h a;
#pragma unroll
    for (int p = 0; p < 8; ++p) {
      int k = ks * 32 + half * 16 + (p << 1);
      a[2 * p]     = (_Float16)trow[k];
      a[2 * p + 1] = (_Float16)trow[k + 1];
    }
#pragma unroll
    for (int nt = 0; nt < NT; ++nt) {
      const v16h b = *(const v16h*)(pack + ((ks * NT + nt) * 256 + lane * 8));
      acc[nt] = __builtin_amdgcn_wmma_f32_16x16x32_f16(
          /*neg_a=*/false, a, /*neg_b=*/false, b,
          /*c_mod=*/(short)0, acc[nt], /*reuse_a=*/false, /*reuse_b=*/false);
    }
  }
}

template<int KD, int NO, int ACT, bool TWO>
__global__ __launch_bounds__(256) void gemm_kernel(const float* __restrict__ X1,
                                                   const float* __restrict__ X2,
                                                   const uint32_t* __restrict__ P1,
                                                   const uint32_t* __restrict__ P2,
                                                   const float* __restrict__ bias,
                                                   float* __restrict__ Y) {
  constexpr int NT = NO / 16;
  constexpr int TSZ = 128 * (KD + 4);
  constexpr int NOPS = (128 * KD) / (256 * 4);   // async ops per thread per tile
  __shared__ float tile[(TWO ? 2 : 1) * TSZ];
  const int tid = threadIdx.x;
  const int lane = tid & 31, wave = tid >> 5;
  const long rowBase = (long)blockIdx.x * 128;

  v8f acc[NT];
#pragma unroll
  for (int i = 0; i < NT; ++i) { v8f z = {}; acc[i] = z; }

  // kick off copies for all phases up front (double-buffered in TWO mode)
  gemm_stage<KD>(tile, X1, rowBase, tid);
  if constexpr (TWO) gemm_stage<KD>(tile + TSZ, X2, rowBase, tid);
  if (blockIdx.x + 1 < gridDim.x)
    __builtin_prefetch(X1 + (rowBase + 128) * KD + tid * 4, 0, 1);  // global_prefetch_b8

  if constexpr (TWO) wait_async<NOPS>();  // buffer 0 landed (async ops complete in order)
  else               wait_async<0>();
  __syncthreads();
  gemm_phase<KD, NT>(tile, P1, lane, wave, acc);
  if constexpr (TWO) {
    wait_async<0>();                      // buffer 1 landed
    __syncthreads();
    gemm_phase<KD, NT>(tile + TSZ, P2, lane, wave, acc);
  }

  // D layout: lane L holds col = L%16; VGPR r -> row r (lanes 0-15) / 8+r (lanes 16-31)
  const int col = lane & 15, half = lane >> 4;
#pragma unroll
  for (int nt = 0; nt < NT; ++nt) {
    const float bv = bias[nt * 16 + col];
#pragma unroll
    for (int r = 0; r < 8; ++r) {
      float v = acc[nt][r] + bv;
      if (ACT == 1) v = fmaxf(v, 0.0f);
      const long row = rowBase + wave * 16 + half * 8 + r;
      Y[row * NO + nt * 16 + col] = v;
    }
  }
}

// ------------------------------------------------------- edge / graph kernels
__global__ void edge_feat_sum_kernel(const float* __restrict__ X, const int* __restrict__ src,
                                     const int* __restrict__ dst, float* __restrict__ out) {
  int t = blockIdx.x * blockDim.x + threadIdx.x;  // E*64 threads, feature-major
  int e = t >> 6, f = t & 63;
  if (e < ETOT)
    atomicAdd(&out[(long)dst[e] * 64 + f], X[(long)src[e] * 64 + f]);
}

__global__ void score_kernel(const float* __restrict__ hs2, const float* __restrict__ ws,
                             const float* __restrict__ bs, float* __restrict__ s) {
  int i = blockIdx.x * blockDim.x + threadIdx.x;
  if (i >= NTOT) return;
  float a = bs[0];
#pragma unroll
  for (int f = 0; f < 32; ++f) a += hs2[i * 32 + f] * ws[f];
  s[i] = tanhf(a);
}

__global__ void mc_kernel(const float* __restrict__ s, const int* __restrict__ src,
                          const int* __restrict__ dst, float* __restrict__ out, float scale) {
  __shared__ float red[256];
  int tid = threadIdx.x;
  int e = blockIdx.x * 256 + tid;
  red[tid] = s[src[e]] * s[dst[e]];   // grid covers E exactly
  __syncthreads();
  for (int st = 128; st > 0; st >>= 1) {
    if (tid < st) red[tid] += red[tid + st];
    __syncthreads();
  }
  if (tid == 0) atomicAdd(out, red[0] * scale);
}

// per-graph descending stable sort of 4096 scores; emit top 2048 as supernodes
__global__ __launch_bounds__(1024) void topk_kernel(const float* __restrict__ s,
                                                    int* __restrict__ perm,
                                                    int* __restrict__ cluster) {
  __shared__ unsigned long long keys[NNODE];
  const int b = blockIdx.x, tid = threadIdx.x;
  for (int i = tid; i < NNODE; i += 1024) {
    unsigned bits = __float_as_uint(s[b * NNODE + i]);
    unsigned u = (bits & 0x80000000u) ? ~bits : (bits | 0x80000000u);  // ascending order key
    keys[i] = ((unsigned long long)(~u) << 32) | (unsigned)i;          // asc sort => desc score, ties: low idx
  }
  for (unsigned k = 2; k <= (unsigned)NNODE; k <<= 1) {
    for (unsigned j = k >> 1; j > 0; j >>= 1) {
      __syncthreads();
      for (unsigned i = tid; i < (unsigned)NNODE; i += 1024) {
        unsigned ixj = i ^ j;
        if (ixj > i) {
          unsigned long long a = keys[i], c = keys[ixj];
          bool up = ((i & k) == 0);
          if ((a > c) == up) { keys[i] = c; keys[ixj] = a; }
        }
      }
    }
  }
  __syncthreads();
  for (int j = tid; j < KPER; j += 1024) {
    int idx = (int)(keys[j] & 0xFFFFFFFFull);
    int node = b * NNODE + idx;
    int cl = b * KPER + j;
    perm[cl] = node;
    cluster[node] = cl;
  }
}

__global__ void gather_xp_kernel(const float* __restrict__ h, const float* __restrict__ s,
                                 const int* __restrict__ perm, float* __restrict__ xp) {
  int t = blockIdx.x * blockDim.x + threadIdx.x;   // KTOT*64
  int k = t >> 6, f = t & 63;
  if (k < KTOT) {
    int node = perm[k];
    xp[t] = h[(long)node * 64 + f] * s[node];
  }
}

__global__ void prop_edge_kernel(const int* __restrict__ src, const int* __restrict__ dst,
                                 const int* __restrict__ cl, int* __restrict__ cand) {
  int e = blockIdx.x * blockDim.x + threadIdx.x;
  if (e < ETOT) {
    int c = cl[src[e]];
    if (c >= 0) atomicMax(&cand[dst[e]], c);
  }
}

__global__ void prop_update_kernel(int* __restrict__ cl, const int* __restrict__ cand) {
  int i = blockIdx.x * blockDim.x + threadIdx.x;
  if (i < NTOT && cl[i] < 0) cl[i] = cand[i];
}

__global__ void cluster_final_kernel(int* __restrict__ cl) {
  int i = blockIdx.x * blockDim.x + threadIdx.x;
  if (i < NTOT && cl[i] < 0) cl[i] = (i >> 12) * KPER;   // graph b -> first supernode
}

// hash-set dedup of coarse edges (ci,cj), ci!=cj; compact unique pairs
__global__ void dedup_kernel(const int* __restrict__ src, const int* __restrict__ dst,
                             const int* __restrict__ cl, unsigned* __restrict__ hashT,
                             int* __restrict__ pairs, unsigned* __restrict__ count) {
  int e = blockIdx.x * blockDim.x + threadIdx.x;
  if (e >= ETOT) return;
  int ci = cl[src[e]], cj = cl[dst[e]];
  if (ci == cj) return;
  unsigned key = (unsigned)ci * (unsigned)KTOT + (unsigned)cj;   // < 2^30
  unsigned slot = (key * 2654435761u) & (HSIZE - 1);
  for (;;) {
    unsigned prev = atomicCAS(&hashT[slot], 0xFFFFFFFFu, key);
    if (prev == 0xFFFFFFFFu) {
      unsigned p = atomicAdd(count, 1u);
      pairs[2 * p] = ci;
      pairs[2 * p + 1] = cj;
      return;
    }
    if (prev == key) return;
    slot = (slot + 1) & (HSIZE - 1);
  }
}

__global__ void coarse_agg_kernel(const int* __restrict__ pairs, const unsigned* __restrict__ count,
                                  const float* __restrict__ xp, float* __restrict__ agg2) {
  int t = blockIdx.x * blockDim.x + threadIdx.x;   // ETOT*64 (upper bound), guarded by *count
  int p = t >> 6, f = t & 63;
  if (p < (int)*count) {
    int ci = pairs[2 * p], cj = pairs[2 * p + 1];
    atomicAdd(&agg2[(long)ci * 64 + f], xp[(long)cj * 64 + f]);
  }
}

__global__ void readout_kernel(const float* __restrict__ h2, float* __restrict__ sums) {
  int t = blockIdx.x * blockDim.x + threadIdx.x;   // 1024 threads: (b,f)
  if (t >= BGRAPH * 64) return;
  int b = t >> 6, f = t & 63;
  float acc = 0.0f;
  for (int r = 0; r < KPER; ++r) acc += h2[((long)(b * KPER + r)) * 64 + f];
  sums[t] = acc * (1.0f / (float)KPER);
}

__global__ void head_kernel(const float* __restrict__ sums, const float* __restrict__ wl,
                            const float* __restrict__ bl, float* __restrict__ out) {
  __shared__ float lg[BGRAPH][CLS];
  __shared__ float mx[BGRAPH];
  __shared__ float ls[BGRAPH];
  int t = threadIdx.x;
  if (t < BGRAPH * CLS) {
    int b = t / CLS, c = t % CLS;
    float a = bl[c];
    for (int f = 0; f < 64; ++f) a += sums[b * 64 + f] * wl[f * CLS + c];
    lg[b][c] = a;
  }
  __syncthreads();
  if (t < BGRAPH) {
    float m = -1e30f;
    for (int c = 0; c < CLS; ++c) m = fmaxf(m, lg[t][c]);
    float se = 0.0f;
    for (int c = 0; c < CLS; ++c) se += expf(lg[t][c] - m);
    mx[t] = m;
    ls[t] = logf(se);
  }
  __syncthreads();
  if (t < BGRAPH * CLS) {
    int b = t / CLS, c = t % CLS;
    out[t] = lg[b][c] - mx[b] - ls[b];
  }
}

}  // namespace

// ---------------------------------------------------------------- launcher
extern "C" void kernel_launch(void* const* d_in, const int* in_sizes, int n_in,
                              void* d_out, int out_size, void* d_ws, size_t ws_size,
                              hipStream_t stream) {
  const float* x   = (const float*)d_in[0];
  const int* ei    = (const int*)d_in[1];
  const int* src   = ei;
  const int* dst   = ei + ETOT;
  const float* w1a = (const float*)d_in[3];  const float* b1a = (const float*)d_in[4];
  const float* w1b = (const float*)d_in[5];  const float* b1b = (const float*)d_in[6];
  const float* wms = (const float*)d_in[7];  const float* wmn = (const float*)d_in[8];
  const float* bm  = (const float*)d_in[9];
  const float* wa  = (const float*)d_in[10]; const float* ba  = (const float*)d_in[11];
  const float* wb  = (const float*)d_in[12]; const float* bb  = (const float*)d_in[13];
  const float* wsc = (const float*)d_in[14]; const float* bs  = (const float*)d_in[15];
  const float* w2a = (const float*)d_in[16]; const float* b2a = (const float*)d_in[17];
  const float* w2b = (const float*)d_in[18]; const float* b2b = (const float*)d_in[19];
  const float* wl  = (const float*)d_in[20]; const float* bl  = (const float*)d_in[21];

  float* out = (float*)d_out;
  float*    wf = (float*)d_ws;
  uint32_t* wu = (uint32_t*)d_ws;

  // workspace layout (dwords); every offset is a multiple of 64 dwords (256B aligned)
  const size_t oA    = 0;
  const size_t oB    = oA + (size_t)NTOT * 64;
  const size_t oC    = oB + (size_t)NTOT * 64;
  const size_t oS    = oC + (size_t)NTOT * 64;
  const size_t oCl   = oS + NTOT;
  const size_t oCd   = oCl + NTOT;
  const size_t oPerm = oCd + NTOT;
  const size_t oSums = oPerm + KTOT;
  const size_t oCnt  = oSums + BGRAPH * 64;
  const size_t oPk   = oCnt + 64;

  float* bufA = wf + oA;   // agg1 -> aggH -> hs1|hs2 -> (hash|pairs)
  float* bufB = wf + oB;   // t1 -> hs0 -> (xp | agg2)
  float* bufC = wf + oC;   // h -> (t2 | h2)
  float* sArr = wf + oS;
  int* cluster   = (int*)(wu + oCl);
  int* cand      = (int*)(wu + oCd);
  int* perm      = (int*)(wu + oPerm);
  float* sums    = wf + oSums;
  unsigned* cnt  = (unsigned*)(wu + oCnt);
  unsigned* hashT = (unsigned*)bufA;
  int* pairs      = (int*)(wu + oA + HSIZE);
  float* xp   = bufB;
  float* agg2 = bufB + (size_t)KTOT * 64;
  float* t2   = bufC;
  float* h2   = bufC + (size_t)KTOT * 64;

  uint32_t* pw1a = wu + oPk;
  uint32_t* pw1b = pw1a + 2048;
  uint32_t* pwms = pw1b + 2048;
  uint32_t* pwmn = pwms + 2048;
  uint32_t* pwa  = pwmn + 2048;  // 1024 dwords
  uint32_t* pwb  = pwa + 1024;   // 512 dwords
  uint32_t* pw2a = pwb + 512;
  uint32_t* pw2b = pw2a + 2048;

  const dim3 blk(256);
  auto nb = [](long n) { return dim3((unsigned)((n + 255) / 256)); };

  // 0) zero output (log-softmax slots fully overwritten; slot 160 accumulates mc_loss)
  fill_u32_kernel<<<nb(BGRAPH * CLS + 1), blk, 0, stream>>>((uint32_t*)out, 0u, BGRAPH * CLS + 1);

  // 1) pack all GEMM weights to f16 B-fragments
  pack_weight_kernel<<<nb(2048), blk, 0, stream>>>(w1a, pw1a, 64, 64);
  pack_weight_kernel<<<nb(2048), blk, 0, stream>>>(w1b, pw1b, 64, 64);
  pack_weight_kernel<<<nb(2048), blk, 0, stream>>>(wms, pwms, 64, 64);
  pack_weight_kernel<<<nb(2048), blk, 0, stream>>>(wmn, pwmn, 64, 64);
  pack_weight_kernel<<<nb(1024), blk, 0, stream>>>(wa,  pwa,  64, 32);
  pack_weight_kernel<<<nb(512),  blk, 0, stream>>>(wb,  pwb,  32, 32);
  pack_weight_kernel<<<nb(2048), blk, 0, stream>>>(w2a, pw2a, 64, 64);
  pack_weight_kernel<<<nb(2048), blk, 0, stream>>>(w2b, pw2b, 64, 64);

  // 2) GINConv1: agg = segsum(x); h = relu((x+agg)@w1a+b1a)@w1b+b1b
  fill_u32_kernel<<<nb((long)NTOT * 64), blk, 0, stream>>>((uint32_t*)bufA, 0u, NTOT * 64);
  edge_feat_sum_kernel<<<nb((long)ETOT * 64), blk, 0, stream>>>(x, src, dst, bufA);
  gemm_kernel<64, 64, 1, true ><<<dim3(NTOT / 128), blk, 0, stream>>>(x,    bufA, pw1a, pw1a, b1a, bufB);
  gemm_kernel<64, 64, 0, false><<<dim3(NTOT / 128), blk, 0, stream>>>(bufB, nullptr, pw1b, nullptr, b1b, bufC);

  // 3) ScoreNet: hs0 = relu(h@wms + segsum(h)@wmn + bm); 2-layer MLP; tanh head
  fill_u32_kernel<<<nb((long)NTOT * 64), blk, 0, stream>>>((uint32_t*)bufA, 0u, NTOT * 64);
  edge_feat_sum_kernel<<<nb((long)ETOT * 64), blk, 0, stream>>>(bufC, src, dst, bufA);
  gemm_kernel<64, 64, 1, true ><<<dim3(NTOT / 128), blk, 0, stream>>>(bufC, bufA, pwms, pwmn, bm, bufB);
  gemm_kernel<64, 32, 1, false><<<dim3(NTOT / 128), blk, 0, stream>>>(bufB, nullptr, pwa, nullptr, ba, bufA);
  gemm_kernel<32, 32, 1, false><<<dim3(NTOT / 128), blk, 0, stream>>>(bufA, nullptr, pwb, nullptr, bb,
                                                                      bufA + (size_t)NTOT * 32);
  score_kernel<<<nb(NTOT), blk, 0, stream>>>(bufA + (size_t)NTOT * 32, wsc, bs, sArr);

  // 4) MaxCut loss: BETA * sum(s[src]*s[dst]) / E  -> out[160]
  mc_kernel<<<dim3(ETOT / 256), blk, 0, stream>>>(sArr, src, dst, out + BGRAPH * CLS, 1.0f / (float)ETOT);

  // 5) per-graph top-k supernodes; xp = h[perm]*s[perm]
  fill_u32_kernel<<<nb(NTOT), blk, 0, stream>>>((uint32_t*)cluster, 0xFFFFFFFFu, NTOT);
  topk_kernel<<<dim3(BGRAPH), dim3(1024), 0, stream>>>(sArr, perm, cluster);
  gather_xp_kernel<<<nb((long)KTOT * 64), blk, 0, stream>>>(bufC, sArr, perm, xp);

  // 6) label propagation (segment-max), 10 rounds + fallback
  for (int it = 0; it < 10; ++it) {
    fill_u32_kernel<<<nb(NTOT), blk, 0, stream>>>((uint32_t*)cand, 0xFFFFFFFFu, NTOT);
    prop_edge_kernel<<<nb(ETOT), blk, 0, stream>>>(src, dst, cluster, cand);
    prop_update_kernel<<<nb(NTOT), blk, 0, stream>>>(cluster, cand);
  }
  cluster_final_kernel<<<nb(NTOT), blk, 0, stream>>>(cluster);

  // 7) coarsen: dedup coarse edges via hash set, then agg2[ci] += xp[cj]
  fill_u32_kernel<<<nb((long)HSIZE), blk, 0, stream>>>(hashT, 0xFFFFFFFFu, (int)HSIZE);
  fill_u32_kernel<<<nb(64), blk, 0, stream>>>((uint32_t*)cnt, 0u, 64);
  fill_u32_kernel<<<nb((long)KTOT * 64), blk, 0, stream>>>((uint32_t*)agg2, 0u, KTOT * 64);
  dedup_kernel<<<nb(ETOT), blk, 0, stream>>>(src, dst, cluster, hashT, pairs, cnt);
  coarse_agg_kernel<<<nb((long)ETOT * 64), blk, 0, stream>>>(pairs, cnt, xp, agg2);

  // 8) GINConv2 on coarse graph
  gemm_kernel<64, 64, 1, true ><<<dim3(KTOT / 128), blk, 0, stream>>>(xp, agg2, pw2a, pw2a, b2a, t2);
  gemm_kernel<64, 64, 0, false><<<dim3(KTOT / 128), blk, 0, stream>>>(t2, nullptr, pw2b, nullptr, b2b, h2);

  // 9) mean readout + classifier + log_softmax
  readout_kernel<<<dim3(4), blk, 0, stream>>>(h2, sums);
  head_kernel<<<dim3(1), blk, 0, stream>>>(sums, wl, bl, out);

  (void)in_sizes; (void)n_in; (void)out_size; (void)ws_size;
}
